// SelfAttention_88252987998915
// MI455X (gfx1250) — matrix operand order repaired and checked by reading, not compiled, not run on hardware
//
#include <hip/hip_runtime.h>

#define NPIX 4096
#define CIN  64
#define CQK  8
#define BATCH 8
#define VPAD 72   // V row stride in halves (144B = 16B-aligned, conflict-free b128 reads)
#define KPAD 24   // K row stride in halves (48B): halves 8..15 zeroed = A-operand pad
#define NBLK (NPIX / 64)
#define LOG2E 1.44269504f

#if __has_builtin(__builtin_amdgcn_global_load_async_to_lds_b128) && \
    __has_builtin(__builtin_amdgcn_s_wait_asynccnt)
#define USE_ASYNC 1
#else
#define USE_ASYNC 0
#endif

typedef __attribute__((ext_vector_type(16))) _Float16 v16h;
typedef __attribute__((ext_vector_type(8)))  float    v8f;
typedef __attribute__((ext_vector_type(2)))  __fp16   v2hf;  // cvt_pkrtz result type
typedef int v4i __attribute__((vector_size(16)));            // builtin's b128 unit

union V16 { v16h v; unsigned w[8]; uint4 q[2]; };

// lane <-> lane^16 swap, pure VALU (v_permlanex16 with identity selects)
__device__ __forceinline__ unsigned swap16u(unsigned x) {
    return (unsigned)__builtin_amdgcn_permlanex16((int)x, (int)x,
                                                  0x76543210, 0xfedcba98,
                                                  true, false);
}
__device__ __forceinline__ float swap16f(float x) {
    return __builtin_bit_cast(float, swap16u(__builtin_bit_cast(unsigned, x)));
}
// pack two f32 -> packed f16 word (v_cvt_pk_rtz_f16_f32)
__device__ __forceinline__ unsigned pk16(float a, float b) {
    v2hf p = __builtin_amdgcn_cvt_pkrtz(a, b);
    return __builtin_bit_cast(unsigned, p);
}

#if USE_ASYNC
typedef __attribute__((address_space(1))) v4i* gv4i_p;   // global (AS1)
typedef __attribute__((address_space(3))) v4i* lv4i_p;   // LDS (AS3)
// 16B per-lane async copy global -> LDS (tracked by ASYNCcnt)
__device__ __forceinline__ void async_cp16(const _Float16* g, _Float16* l) {
    __builtin_amdgcn_global_load_async_to_lds_b128(
        (gv4i_p)(v4i*)g, (lv4i_p)(v4i*)l, 0, 0);
}
#endif

// ---------------------------------------------------------------------------
// Stage 1: 1x1-conv projections -> f16 workspace
//   Kw (=fx): (B, N, 8), Qw (=gx^T): (B, N, 8), Vw (=hx): (B, 64, N)
// ---------------------------------------------------------------------------
__global__ __launch_bounds__(256) void proj_kernel(
    const float* __restrict__ x,
    const float* __restrict__ Wf, const float* __restrict__ bf,
    const float* __restrict__ Wg, const float* __restrict__ bg,
    const float* __restrict__ Wh, const float* __restrict__ bh,
    _Float16* __restrict__ Kw, _Float16* __restrict__ Qw,
    _Float16* __restrict__ Vw)
{
    __shared__ __attribute__((aligned(16))) float WfL[CQK*CIN];
    __shared__ __attribute__((aligned(16))) float WgL[CQK*CIN];
    __shared__ __attribute__((aligned(16))) float WhL[CIN*CIN];
    __shared__ float bfL[CQK], bgL[CQK], bhL[CIN];

    int tid = threadIdx.x;
    for (int i = tid; i < CQK*CIN; i += 256) { WfL[i] = Wf[i]; WgL[i] = Wg[i]; }
    for (int i = tid; i < CIN*CIN; i += 256) WhL[i] = Wh[i];
    if (tid < CQK) { bfL[tid] = bf[tid]; bgL[tid] = bg[tid]; }
    if (tid < CIN) bhL[tid] = bh[tid];
    __syncthreads();

    int p = blockIdx.x * 256 + tid;      // global pixel id: b*4096 + n
    int b = p >> 12;
    int n = p & (NPIX - 1);

    float aF[CQK], aG[CQK], aH[CIN];
#pragma unroll
    for (int o = 0; o < CQK; o++) { aF[o] = bfL[o]; aG[o] = bgL[o]; }
#pragma unroll
    for (int o = 0; o < CIN; o++) aH[o] = bhL[o];

    const float* xb = x + (size_t)b * CIN * NPIX + n;
    for (int c = 0; c < CIN; c += 4) {
        float x0 = xb[(size_t)(c+0)*NPIX];
        float x1 = xb[(size_t)(c+1)*NPIX];
        float x2 = xb[(size_t)(c+2)*NPIX];
        float x3 = xb[(size_t)(c+3)*NPIX];
#pragma unroll
        for (int o = 0; o < CQK; o++) {
            float4 wf = *(const float4*)&WfL[o*CIN + c];
            aF[o] = fmaf(x0, wf.x, fmaf(x1, wf.y, fmaf(x2, wf.z, fmaf(x3, wf.w, aF[o]))));
            float4 wg = *(const float4*)&WgL[o*CIN + c];
            aG[o] = fmaf(x0, wg.x, fmaf(x1, wg.y, fmaf(x2, wg.z, fmaf(x3, wg.w, aG[o]))));
        }
#pragma unroll
        for (int o = 0; o < CIN; o++) {
            float4 wh = *(const float4*)&WhL[o*CIN + c];
            aH[o] = fmaf(x0, wh.x, fmaf(x1, wh.y, fmaf(x2, wh.z, fmaf(x3, wh.w, aH[o]))));
        }
    }

    uint4 pk;
    pk.x = pk16(aF[0], aF[1]); pk.y = pk16(aF[2], aF[3]);
    pk.z = pk16(aF[4], aF[5]); pk.w = pk16(aF[6], aF[7]);
    *(uint4*)(Kw + (size_t)p * CQK) = pk;
    pk.x = pk16(aG[0], aG[1]); pk.y = pk16(aG[2], aG[3]);
    pk.z = pk16(aG[4], aG[5]); pk.w = pk16(aG[6], aG[7]);
    *(uint4*)(Qw + (size_t)p * CQK) = pk;

    _Float16* vrow = Vw + (size_t)b * CIN * NPIX + n;
#pragma unroll
    for (int o = 0; o < CIN; o++) vrow[(size_t)o * NPIX] = (_Float16)aH[o];
}

// ---------------------------------------------------------------------------
// Stage 2: flash attention. 1 wave32 = 16 queries; 64-key blocks;
// 12 v_wmma_f32_16x16x32_f16 per block; permlanex16 for half-wave exchange;
// double-buffered async global->LDS staging; A-operand zero-padding in LDS.
// ---------------------------------------------------------------------------
__global__ __launch_bounds__(256) void attn_kernel(
    const float* __restrict__ x, const float* __restrict__ gamma,
    const _Float16* __restrict__ Kw, const _Float16* __restrict__ Qw,
    const _Float16* __restrict__ Vw, float* __restrict__ out)
{
    __shared__ __attribute__((aligned(16))) _Float16 Ks[2][64 * KPAD];
    __shared__ __attribute__((aligned(16))) _Float16 Vs[2][CIN * VPAD];
    __shared__ __attribute__((aligned(16))) _Float16 Kz[64 * KPAD];  // all-zero A rows

    int tid  = threadIdx.x;
    int lane = tid & 31;
    int wave = tid >> 5;
    int b    = blockIdx.x >> 5;              // 32 query tiles per batch
    int tile = blockIdx.x & 31;
    int m0   = tile * 128 + wave * 16;       // this wave's first query
    bool lo  = lane < 16;
    int  l16 = lane & 15;
    const uint4 z4 = make_uint4(0, 0, 0, 0);

    // zero-init: per-row A-operand pad (halves 8..15) in both K buffers + Kz
    for (int i = tid; i < 128; i += 256)
        *(uint4*)&Ks[i >> 6][(i & 63) * KPAD + 8] = z4;
    for (int i = tid; i < (64 * KPAD) / 8; i += 256)
        *(uint4*)&Kz[i * 8] = z4;

    // Q as WMMA B matrix (32x16): lanes 0-15 = K-rows 0-15 (head 0-7, zero pad),
    // lanes 16-31 = K-rows 16-31 (all zero: head dim is 8).
    v16h bq;
    {
        V16 u;
        uint4 qv = *(const uint4*)(Qw + (size_t)(b * NPIX + m0 + l16) * CQK);
        u.q[0] = lo ? qv : z4;
        u.q[1] = z4;
        bq = u.v;
    }

    v8f acc[4];
#pragma unroll
    for (int g = 0; g < 4; g++) { v8f z = {}; acc[g] = z; }

    float m_run = -1e30f;
    float l_run = 0.0f;

    const _Float16* Kb = Kw + (size_t)b * NPIX * CQK;
    const _Float16* Vb = Vw + (size_t)b * CIN * NPIX;

    // cooperative staging roles
    int  vch  = tid >> 2;            // 0..63 : value channel
    int  vs0  = (tid & 3) * 16;      // 0,16,32,48 : key sub-segment (halves)
    bool kldr = tid < 64;            // waves 0,1 stage K (wave-uniform)

    // 64-key block: 4 score WMMA + online softmax + 8 PV WMMA
    auto compute_block = [&](const _Float16* KsC, const _Float16* VsC) {
        // ---- scores: A operands straight from LDS (pad pre-zeroed) ----
        const _Float16* abase = lo ? KsC : Kz;
        V16 ua[4];
#pragma unroll
        for (int c = 0; c < 4; c++) {
            const _Float16* r = abase + (c * 16 + l16) * KPAD;
            ua[c].q[0] = *(const uint4*)(r);
            ua[c].q[1] = *(const uint4*)(r + 8);
        }
        v8f d[4];
#pragma unroll
        for (int c = 0; c < 4; c++) {
            v8f z = {};
            d[c] = __builtin_amdgcn_wmma_f32_16x16x32_f16(false, ua[c].v, false, bq,
                                                          (short)0, z, false, false);
        }

        // ---- online softmax (tree reductions; lanes j and j+16 share query j) ----
        float mm[8];
#pragma unroll
        for (int i = 0; i < 8; i++)
            mm[i] = fmaxf(fmaxf(d[0][i], d[1][i]), fmaxf(d[2][i], d[3][i]));
        float lm = fmaxf(fmaxf(fmaxf(mm[0], mm[1]), fmaxf(mm[2], mm[3])),
                         fmaxf(fmaxf(mm[4], mm[5]), fmaxf(mm[6], mm[7])));
        lm = fmaxf(lm, swap16f(lm));
        float nm = fmaxf(m_run, lm);
        if (__any(nm > m_run)) {           // rescale only when some query's max moved
            float sc = exp2f((m_run - nm) * LOG2E);
#pragma unroll
            for (int g = 0; g < 4; g++)
#pragma unroll
                for (int r = 0; r < 8; r++) acc[g][r] *= sc;
            l_run *= sc;
            m_run = nm;
        }
        float nl = m_run * LOG2E;

        unsigned P[4][4];                 // packed f16 exp pairs per chunk
        float cs[4];
#pragma unroll
        for (int c = 0; c < 4; c++) {
            float e[8];
#pragma unroll
            for (int i = 0; i < 8; i++) e[i] = exp2f(fmaf(d[c][i], LOG2E, -nl));
#pragma unroll
            for (int v = 0; v < 4; v++) P[c][v] = pk16(e[2*v], e[2*v+1]);
            cs[c] = ((e[0]+e[1]) + (e[2]+e[3])) + ((e[4]+e[5]) + (e[6]+e[7]));
        }
        float ls = (cs[0] + cs[1]) + (cs[2] + cs[3]);
        ls += swap16f(ls);
        l_run += ls;

        // ---- P -> B-matrix layout for two 32-key groups ----
        V16 bp[2];
#pragma unroll
        for (int v = 0; v < 4; v++) {
            unsigned s0 = lo ? P[1][v] : P[0][v];
            unsigned t0 = swap16u(s0);
            bp[0].w[v]     = lo ? P[0][v] : t0;
            bp[0].w[4 + v] = lo ? t0 : P[1][v];
            unsigned s1 = lo ? P[3][v] : P[2][v];
            unsigned t1 = swap16u(s1);
            bp[1].w[v]     = lo ? P[2][v] : t1;
            bp[1].w[4 + v] = lo ? t1 : P[3][v];
        }

        // ---- O += V * P : batched V loads, then 4 back-to-back WMMAs ----
        int hb = lo ? 0 : 16;
#pragma unroll
        for (int half = 0; half < 2; half++) {
            V16 av[4];
#pragma unroll
            for (int g = 0; g < 4; g++) {
                const _Float16* vrow = VsC + (g * 16 + l16) * VPAD + half * 32 + hb;
                av[g].q[0] = *(const uint4*)(vrow);
                av[g].q[1] = *(const uint4*)(vrow + 8);
            }
#pragma unroll
            for (int g = 0; g < 4; g++)
                acc[g] = __builtin_amdgcn_wmma_f32_16x16x32_f16(false, av[g].v, false,
                                                                bp[half].v, (short)0,
                                                                acc[g], false, false);
        }
    };

#if USE_ASYNC
    auto stage = [&](int buf, int n0) {
        const _Float16* vsrc = Vb + (size_t)vch * NPIX + n0 + vs0;
        async_cp16(vsrc,     &Vs[buf][vch * VPAD + vs0]);
        async_cp16(vsrc + 8, &Vs[buf][vch * VPAD + vs0 + 8]);
        if (kldr) async_cp16(Kb + (size_t)(n0 + tid) * CQK, &Ks[buf][tid * KPAD]);
    };
    stage(0, 0);
    for (int i = 0; i < NBLK; i++) {
        __builtin_amdgcn_s_wait_asynccnt(0);
        __syncthreads();
        int nx = (i + 1 < NBLK) ? i + 1 : i;
        stage((i + 1) & 1, nx * 64);
        compute_block(Ks[i & 1], Vs[i & 1]);
    }
#else
    // synchronous register-pipelined fallback
    uint4 rv0, rv1, rk = z4;
    {
        const _Float16* vsrc = Vb + (size_t)vch * NPIX + vs0;
        rv0 = *(const uint4*)(vsrc);
        rv1 = *(const uint4*)(vsrc + 8);
        if (kldr) rk = *(const uint4*)(Kb + (size_t)tid * CQK);
    }
    for (int i = 0; i < NBLK; i++) {
        __syncthreads();
        *(uint4*)&Vs[0][vch * VPAD + vs0]     = rv0;
        *(uint4*)&Vs[0][vch * VPAD + vs0 + 8] = rv1;
        if (kldr) *(uint4*)&Ks[0][tid * KPAD] = rk;
        __syncthreads();
        int nx = (i + 1 < NBLK) ? i + 1 : i;
        const _Float16* vsrc = Vb + (size_t)vch * NPIX + nx * 64 + vs0;
        rv0 = *(const uint4*)(vsrc);
        rv1 = *(const uint4*)(vsrc + 8);
        if (kldr) rk = *(const uint4*)(Kb + (size_t)(nx * 64 + tid) * CQK);
        compute_block(Ks[0], Vs[0]);
    }
#endif

    // ---- epilogue: out = x + gamma * O / l ----
    float inv_l = 1.0f / l_run;
    float gm = gamma[0];
    int m  = m0 + l16;
    int cb = lo ? 0 : 8;
#pragma unroll
    for (int g = 0; g < 4; g++)
#pragma unroll
        for (int r = 0; r < 8; r++) {
            int c = g * 16 + cb + r;
            size_t idx = ((size_t)(b * CIN + c)) * NPIX + m;
            out[idx] = x[idx] + gm * acc[g][r] * inv_l;
        }
}

// ---------------------------------------------------------------------------
extern "C" void kernel_launch(void* const* d_in, const int* in_sizes, int n_in,
                              void* d_out, int out_size, void* d_ws, size_t ws_size,
                              hipStream_t stream) {
    const float* x     = (const float*)d_in[0];
    const float* Wf    = (const float*)d_in[1];
    const float* bf    = (const float*)d_in[2];
    const float* Wg    = (const float*)d_in[3];
    const float* bg    = (const float*)d_in[4];
    const float* Wh    = (const float*)d_in[5];
    const float* bh    = (const float*)d_in[6];
    const float* gamma = (const float*)d_in[7];
    float* out = (float*)d_out;

    _Float16* Kw = (_Float16*)d_ws;                        // (B, N, 8)   512 KB
    _Float16* Qw = Kw + (size_t)BATCH * NPIX * CQK;        // (B, N, 8)   512 KB
    _Float16* Vw = Qw + (size_t)BATCH * NPIX * CQK;        // (B, 64, N)  4 MB

    proj_kernel<<<(BATCH * NPIX) / 256, 256, 0, stream>>>(
        x, Wf, bf, Wg, bg, Wh, bh, Kw, Qw, Vw);
    attn_kernel<<<BATCH * (NPIX / 128), 256, 0, stream>>>(
        x, gamma, Kw, Qw, Vw, out);
}
